// Head_4011499454905
// MI455X (gfx1250) — matrix operand order repaired and checked by reading, not compiled, not run on hardware
//
#include <hip/hip_runtime.h>
#include <hip/hip_bf16.h>

typedef __attribute__((ext_vector_type(16))) __bf16 v16bf;
typedef __attribute__((ext_vector_type(8)))  __bf16 v8bf;
typedef __attribute__((ext_vector_type(8)))  float  v8f;

#define B_ 4
#define T_ 4096
#define C_ 1024
#define H_ 128

#define WMMA_BF16(A, Bm, Cm) \
  __builtin_amdgcn_wmma_f32_16x16x32_bf16(false, (A), false, (Bm), (short)0, (Cm), false, false)

static __device__ __forceinline__ v16bf mk16(v8bf lo, v8bf hi) {
  v16bf r;
#pragma unroll
  for (int i = 0; i < 8; ++i) { r[i] = lo[i]; r[i + 8] = hi[i]; }
  return r;
}
static __device__ __forceinline__ v8f zero8() {
  v8f z;
#pragma unroll
  for (int i = 0; i < 8; ++i) z[i] = 0.0f;
  return z;
}
static __device__ __forceinline__ float redmax16(float x) {
  x = fmaxf(x, __shfl_xor(x, 1, 32));
  x = fmaxf(x, __shfl_xor(x, 2, 32));
  x = fmaxf(x, __shfl_xor(x, 4, 32));
  x = fmaxf(x, __shfl_xor(x, 8, 32));
  return x;
}
static __device__ __forceinline__ float redsum16(float x) {
  x += __shfl_xor(x, 1, 32);
  x += __shfl_xor(x, 2, 32);
  x += __shfl_xor(x, 4, 32);
  x += __shfl_xor(x, 8, 32);
  return x;
}

// ---- Wq/Wk/Wv (C x H, f32 row-major) -> Wt[3][H][C] bf16 (transposed) ----
__global__ void wtrans_kernel(const float* __restrict__ Wq, const float* __restrict__ Wk,
                              const float* __restrict__ Wv, __bf16* __restrict__ Wt) {
  int tid = blockIdx.x * blockDim.x + threadIdx.x;
  if (tid >= 3 * C_ * H_) return;
  int hcol = tid & (H_ - 1);
  int c = (tid >> 7) & (C_ - 1);
  int m = tid >> 17;
  const float* W = (m == 0) ? Wq : ((m == 1) ? Wk : Wv);
  Wt[(size_t)(m * H_ + hcol) * C_ + c] = (__bf16)W[c * H_ + hcol];
}

// ---- QKV projection: block = 16 rows, 8 waves; ping-pong double buffer ----
__global__ __launch_bounds__(256) void qkv_kernel(const float* __restrict__ x,
                                                  const __bf16* __restrict__ Wt,
                                                  __bf16* __restrict__ Q,
                                                  __bf16* __restrict__ K,
                                                  __bf16* __restrict__ Vt) {
  int blk = blockIdx.x;
  int b = blk / (T_ / 16);
  int t0 = (blk % (T_ / 16)) * 16;
  int w = threadIdx.x >> 5;     // column tile 0..7
  int lane = threadIdx.x & 31;
  int n = lane & 15;
  int h = lane >> 4;

  const float* xrow = x + (size_t)(b * T_ + t0 + n) * C_;
  const __bf16* w0 = Wt + (size_t)(0 * H_ + w * 16 + n) * C_;
  const __bf16* w1 = Wt + (size_t)(1 * H_ + w * 16 + n) * C_;
  const __bf16* w2 = Wt + (size_t)(2 * H_ + w * 16 + n) * C_;

  auto loadA = [&](int k0) -> v16bf {
    // A layout: lane=row, elems 0..7 = K[k0+8h..], elems 8..15 = K[k0+16+8h..]
    const float4 f0 = *(const float4*)(xrow + k0 + 8 * h);
    const float4 f1 = *(const float4*)(xrow + k0 + 8 * h + 4);
    const float4 g0 = *(const float4*)(xrow + k0 + 16 + 8 * h);
    const float4 g1 = *(const float4*)(xrow + k0 + 16 + 8 * h + 4);
    v16bf a;
    a[0] = (__bf16)f0.x;  a[1] = (__bf16)f0.y;  a[2] = (__bf16)f0.z;  a[3] = (__bf16)f0.w;
    a[4] = (__bf16)f1.x;  a[5] = (__bf16)f1.y;  a[6] = (__bf16)f1.z;  a[7] = (__bf16)f1.w;
    a[8] = (__bf16)g0.x;  a[9] = (__bf16)g0.y;  a[10] = (__bf16)g0.z; a[11] = (__bf16)g0.w;
    a[12] = (__bf16)g1.x; a[13] = (__bf16)g1.y; a[14] = (__bf16)g1.z; a[15] = (__bf16)g1.w;
    return a;
  };
  auto loadB = [&](const __bf16* wp, int k0) -> v16bf {
    // B layout: lane=col, elems 0..7 = K[k0+8h..], elems 8..15 = K[k0+16+8h..]
    int kb = k0 + 8 * h;
    return mk16(*(const v8bf*)(wp + kb), *(const v8bf*)(wp + kb + 16));
  };

  v8f acc0 = zero8(), acc1 = zero8(), acc2 = zero8();

  // ping-pong buffers: set A covers k0, set B covers k0+32; no rotation movs
  v16bf aA = loadA(0);
  v16bf b0A = loadB(w0, 0), b1A = loadB(w1, 0), b2A = loadB(w2, 0);

#pragma unroll 1
  for (int k0 = 0; k0 < C_; k0 += 64) {
    v16bf aB  = loadA(k0 + 32);
    v16bf b0B = loadB(w0, k0 + 32);
    v16bf b1B = loadB(w1, k0 + 32);
    v16bf b2B = loadB(w2, k0 + 32);
    acc0 = WMMA_BF16(aA, b0A, acc0);
    acc1 = WMMA_BF16(aA, b1A, acc1);
    acc2 = WMMA_BF16(aA, b2A, acc2);
    if (k0 + 64 < C_) {
      aA  = loadA(k0 + 64);
      b0A = loadB(w0, k0 + 64);
      b1A = loadB(w1, k0 + 64);
      b2A = loadB(w2, k0 + 64);
    }
    acc0 = WMMA_BF16(aB, b0B, acc0);
    acc1 = WMMA_BF16(aB, b1B, acc1);
    acc2 = WMMA_BF16(aB, b2B, acc2);
  }

  // D layout: lane=col n, VGPR v = row v + 8h
#pragma unroll
  for (int v = 0; v < 8; ++v) {
    int t = t0 + v + 8 * h;
    Q[(size_t)(b * T_ + t) * H_ + w * 16 + n] = (__bf16)acc0[v];
    K[(size_t)(b * T_ + t) * H_ + w * 16 + n] = (__bf16)acc1[v];
  }
  v8bf vv;
#pragma unroll
  for (int v = 0; v < 8; ++v) vv[v] = (__bf16)acc2[v];
  *(v8bf*)(Vt + (size_t)(b * H_ + w * 16 + n) * T_ + t0 + 8 * h) = vv;  // V^T
}

// ---- Flash attention: block = 128 query rows (8 waves x 16), K/V chunks
// ---- staged once per block into double-buffered LDS; causal 32-key chunks.
__global__ __launch_bounds__(256) void attn_kernel(const __bf16* __restrict__ Q,
                                                   const __bf16* __restrict__ Kg,
                                                   const __bf16* __restrict__ Vtg,
                                                   float* __restrict__ out) {
  int bidx = blockIdx.x;                 // B * (T/128)
  int b = bidx / (T_ / 128);
  int tb0 = (bidx % (T_ / 128)) * 128;
  int tid = threadIdx.x;
  int w = tid >> 5;                      // wave id: q-tile tb0 + 16*w
  int lane = tid & 31;
  int n = lane & 15;
  int h = lane >> 4;
  int t0w = tb0 + 16 * w;
  const float scale = 0.03125f;          // 1/sqrt(1024)

  __shared__ __align__(16) __bf16 Kc[2][32 * 128];   // [key][d]
  __shared__ __align__(16) __bf16 Vc[2][128 * 32];   // [d][key]  (from V^T)
  __shared__ __align__(16) __bf16 pbuf[8][16 * 32];  // per-wave P transpose

  // Q tile resident in registers (4 A-frags over K=128)
  v16bf aq[4];
  {
    const __bf16* qrow = Q + (size_t)(b * T_ + t0w + n) * H_;
#pragma unroll
    for (int kc = 0; kc < 4; ++kc) {
      int ko = kc * 32 + 8 * h;
      aq[kc] = mk16(*(const v8bf*)(qrow + ko), *(const v8bf*)(qrow + ko + 16));
    }
  }

  v8f acc[8];
#pragma unroll
  for (int d = 0; d < 8; ++d) acc[d] = zero8();
  float mrow[8], lrow[8];
#pragma unroll
  for (int v = 0; v < 8; ++v) { mrow[v] = -1e30f; lrow[v] = 0.0f; }

  // staging address pieces (256 threads move 8KB K + 8KB V per chunk)
  int krow = tid >> 4, kseg = tid & 15;   // K: 16 threads/row, rows 0..15 (+16)
  int vrow = tid >> 2, vseg = tid & 3;    // V: 4 threads/row, rows 0..63 (+64)
  const __bf16* kbase = Kg  + (size_t)(b * T_) * H_;
  const __bf16* vbase = Vtg + (size_t)(b * H_) * T_;

  int nch = tb0 / 32 + 4;                 // chunks cover s <= tb0+127

  // prologue: stage chunk 0 into buffer 0
  {
    v8bf k0r = *(const v8bf*)(kbase + (size_t)(krow) * H_ + kseg * 8);
    v8bf k1r = *(const v8bf*)(kbase + (size_t)(krow + 16) * H_ + kseg * 8);
    v8bf v0r = *(const v8bf*)(vbase + (size_t)(vrow) * T_ + vseg * 8);
    v8bf v1r = *(const v8bf*)(vbase + (size_t)(vrow + 64) * T_ + vseg * 8);
    *(v8bf*)&Kc[0][(krow) * 128 + kseg * 8]      = k0r;
    *(v8bf*)&Kc[0][(krow + 16) * 128 + kseg * 8] = k1r;
    *(v8bf*)&Vc[0][(vrow) * 32 + vseg * 8]       = v0r;
    *(v8bf*)&Vc[0][(vrow + 64) * 32 + vseg * 8]  = v1r;
  }
  __syncthreads();

  for (int ch = 0; ch < nch; ++ch) {
    int cur = ch & 1;
    int s0 = ch * 32;
    bool have_next = (ch + 1 < nch);

    // issue next chunk's global loads now; they fly during compute
    v8bf k0r, k1r, v0r, v1r;
    if (have_next) {
      int s0n = s0 + 32;
      k0r = *(const v8bf*)(kbase + (size_t)(s0n + krow) * H_ + kseg * 8);
      k1r = *(const v8bf*)(kbase + (size_t)(s0n + krow + 16) * H_ + kseg * 8);
      v0r = *(const v8bf*)(vbase + (size_t)(vrow) * T_ + s0n + vseg * 8);
      v1r = *(const v8bf*)(vbase + (size_t)(vrow + 64) * T_ + s0n + vseg * 8);
    }

    if (s0 <= t0w + 15) {   // causal: this wave still has keys in this chunk
      v8f sa = zero8(), sb = zero8();
#pragma unroll
      for (int kc = 0; kc < 4; ++kc) {
        int kb = kc * 32 + 8 * h;
        v16bf b0 = mk16(*(const v8bf*)&Kc[cur][(n) * 128 + kb],
                        *(const v8bf*)&Kc[cur][(n) * 128 + kb + 16]);
        v16bf b1 = mk16(*(const v8bf*)&Kc[cur][(16 + n) * 128 + kb],
                        *(const v8bf*)&Kc[cur][(16 + n) * 128 + kb + 16]);
        sa = WMMA_BF16(aq[kc], b0, sa);
        sb = WMMA_BF16(aq[kc], b1, sb);
      }

      bool needmask = (s0 + 31 > t0w);
      float alpha[8], p0[8], p1[8];
#pragma unroll
      for (int v = 0; v < 8; ++v) {
        float sv0 = sa[v] * scale, sv1 = sb[v] * scale;
        if (needmask) {
          int t = t0w + v + 8 * h;
          if (s0 + n > t)      sv0 = -1e30f;
          if (s0 + 16 + n > t) sv1 = -1e30f;
        }
        float rm = redmax16(fmaxf(sv0, sv1));
        float mnew = fmaxf(mrow[v], rm);
        float al = __expf(mrow[v] - mnew);
        mrow[v] = mnew;
        float e0 = __expf(sv0 - mnew);
        float e1 = __expf(sv1 - mnew);
        float rs = redsum16(e0 + e1);
        lrow[v] = lrow[v] * al + rs;
        alpha[v] = al; p0[v] = e0; p1[v] = e1;
      }
#pragma unroll
      for (int d = 0; d < 8; ++d)
#pragma unroll
        for (int v = 0; v < 8; ++v) acc[d][v] *= alpha[v];

      // transpose P (D layout -> A layout) via per-wave LDS tile.
      // LDS ops are in-order within a wave (DScnt), no block barrier needed.
#pragma unroll
      for (int v = 0; v < 8; ++v) {
        int row = v + 8 * h;
        pbuf[w][row * 32 + n]      = (__bf16)p0[v];
        pbuf[w][row * 32 + 16 + n] = (__bf16)p1[v];
      }
      v16bf pa = mk16(*(const v8bf*)&pbuf[w][n * 32 + 8 * h],
                      *(const v8bf*)&pbuf[w][n * 32 + 16 + 8 * h]);

#pragma unroll
      for (int d = 0; d < 8; ++d) {
        int vb = (d * 16 + n) * 32 + 8 * h;
        v16bf bv = mk16(*(const v8bf*)&Vc[cur][vb], *(const v8bf*)&Vc[cur][vb + 16]);
        acc[d] = WMMA_BF16(pa, bv, acc[d]);
      }
    }

    // drain next chunk's loads into the other LDS buffer (its previous
    // readers finished before the barrier that ended iteration ch-1)
    if (have_next) {
      int nb = cur ^ 1;
      *(v8bf*)&Kc[nb][(krow) * 128 + kseg * 8]      = k0r;
      *(v8bf*)&Kc[nb][(krow + 16) * 128 + kseg * 8] = k1r;
      *(v8bf*)&Vc[nb][(vrow) * 32 + vseg * 8]       = v0r;
      *(v8bf*)&Vc[nb][(vrow + 64) * 32 + vseg * 8]  = v1r;
    }
    __syncthreads();
  }

#pragma unroll
  for (int v = 0; v < 8; ++v) {
    float inv = 1.0f / lrow[v];
    int t = t0w + v + 8 * h;
    float* orow = out + (size_t)(b * T_ + t) * H_;
#pragma unroll
    for (int d = 0; d < 8; ++d) orow[d * 16 + n] = acc[d][v] * inv;
  }
}

extern "C" void kernel_launch(void* const* d_in, const int* in_sizes, int n_in,
                              void* d_out, int out_size, void* d_ws, size_t ws_size,
                              hipStream_t stream) {
  (void)in_sizes; (void)n_in; (void)out_size; (void)ws_size;
  const float* x  = (const float*)d_in[0];
  const float* Wq = (const float*)d_in[1];
  const float* Wk = (const float*)d_in[2];
  const float* Wv = (const float*)d_in[3];
  float* out = (float*)d_out;

  char* ws = (char*)d_ws;
  size_t off = 0;
  __bf16* Wt = (__bf16*)(ws + off); off += (size_t)3 * H_ * C_ * 2;
  __bf16* Qb = (__bf16*)(ws + off); off += (size_t)B_ * T_ * H_ * 2;
  __bf16* Kb = (__bf16*)(ws + off); off += (size_t)B_ * T_ * H_ * 2;
  __bf16* Vt = (__bf16*)(ws + off); off += (size_t)B_ * T_ * H_ * 2;

  wtrans_kernel<<<(3 * C_ * H_ + 255) / 256, 256, 0, stream>>>(Wq, Wk, Wv, Wt);
  qkv_kernel<<<B_ * (T_ / 16), 256, 0, stream>>>(x, Wt, Qb, Kb, Vt);
  attn_kernel<<<B_ * (T_ / 128), 256, 0, stream>>>(Qb, Kb, Vt, out);
}
